// TransformerEdgeClassifier_49795850830537
// MI455X (gfx1250) — matrix operand order repaired and checked by reading, not compiled, not run on hardware
//
#include <hip/hip_runtime.h>
#include <hip/hip_bf16.h>

#define NN   50000
#define EE   500000
#define EMBD 128

typedef __attribute__((ext_vector_type(16))) __bf16 v16bf;
typedef __attribute__((ext_vector_type(8)))  __bf16 v8bf;
typedef __attribute__((ext_vector_type(8)))  float  v8f;

static __device__ __forceinline__ v8f wmma_bf16(v16bf a, v16bf b, v8f c) {
    // D = A(16x32) * B(32x16) + C, bf16 in / f32 out
    return __builtin_amdgcn_wmma_f32_16x16x32_bf16(false, a, false, b, (short)0, c,
                                                   false, false);
}

// -------------------- embedding gather: h = emb[x], + bf16 copy --------------------
__global__ void k_embed(const int* __restrict__ x, const float* __restrict__ emb,
                        float* __restrict__ h, __bf16* __restrict__ hb) {
    int t = blockIdx.x * blockDim.x + threadIdx.x;
    if (t >= NN * EMBD) return;
    int node = t >> 7, c = t & 127;
    float v = emb[x[node] * EMBD + c];
    h[t]  = v;
    hb[t] = (__bf16)v;
}

// -------------------- pack W[K,Nout] (f32, row major) into CDNA5 bf16 B-fragment layout ----
// Fragment tile (kt,nt): 32 lanes x 16 contiguous bf16.
//   lane n<16  : column nt*16+n, K = kt*32 + {0..7, 16..23}
//   lane n>=16 : column nt*16+(n-16), K = kt*32 + {8..15, 24..31}
__global__ void k_pack_b(const float* __restrict__ W, __bf16* __restrict__ out,
                         int K, int Nout) {
    int t = blockIdx.x * blockDim.x + threadIdx.x;
    if (t >= K * Nout) return;
    int k = t / Nout, n = t % Nout;
    int kt = k >> 5, kk = k & 31, nt = n >> 4, nn = n & 15;
    int lane = ((kk >> 3) & 1) ? (16 + nn) : nn;
    int i = (kk & 7) + ((kk >> 4) << 3);
    out[(((size_t)(kt * (Nout >> 4) + nt) * 32) + lane) * 16 + i] = (__bf16)W[t];
}

// -------------------- WMMA GEMM: C[nrows,Nout] = A[nrows,K]*B + bias --------------------
// One wave per 16x32 output tile: TWO independent accumulators share one A fragment,
// so the scheduler interleaves the two WMMA chains across the bf16-WMMA hazard window
// (no dead v_nops) and A-fragment traffic is halved.
__global__ void k_gemm_bf16(const __bf16* __restrict__ A, const __bf16* __restrict__ Bp,
                            const float* __restrict__ bias, float* __restrict__ Cout,
                            int nrows, int K, int Nout) {
    int wave = blockIdx.x * (blockDim.x >> 5) + (threadIdx.x >> 5);
    int lane = threadIdx.x & 31;
    int nColTiles = Nout >> 4;
    int nPair     = nColTiles >> 1;             // Nout is a multiple of 32 here
    int nTiles    = (nrows >> 4) * nPair;
    if (wave >= nTiles) return;                 // wave-uniform: EXEC stays all-ones
    int rowTile = wave / nPair, cp = wave % nPair;
    int ct0 = cp * 2, ct1 = ct0 + 1;

    const __bf16* arow = A + (size_t)((rowTile << 4) + (lane & 15)) * K;
    int kHalf = (lane < 16) ? 0 : 8;            // ISA 16-bit A layout, lane halves
    v8f c0 = {0.f, 0.f, 0.f, 0.f, 0.f, 0.f, 0.f, 0.f};
    v8f c1 = {0.f, 0.f, 0.f, 0.f, 0.f, 0.f, 0.f, 0.f};
    int nkt = K >> 5;
    for (int kt = 0; kt < nkt; ++kt) {
        int k0 = (kt << 5) + kHalf;
        union { v16bf v; v8bf h[2]; } ua;
        ua.h[0] = *(const v8bf*)(arow + k0);        // K = {0..7}/{8..15}
        ua.h[1] = *(const v8bf*)(arow + k0 + 16);   // K = {16..23}/{24..31}
        v16bf b0 = *(const v16bf*)(Bp + (((size_t)(kt * nColTiles + ct0) * 32) + lane) * 16);
        v16bf b1 = *(const v16bf*)(Bp + (((size_t)(kt * nColTiles + ct1) * 32) + lane) * 16);
        c0 = wmma_bf16(ua.v, b0, c0);               // independent chains -> co-issueable
        c1 = wmma_bf16(ua.v, b1, c1);
    }
    int nn    = lane & 15;
    int col0  = (ct0 << 4) + nn;
    int col1  = (ct1 << 4) + nn;
    float bc0 = bias ? bias[col0] : 0.f;
    float bc1 = bias ? bias[col1] : 0.f;
    int rbase = (rowTile << 4) + ((lane >= 16) ? 8 : 0);
#pragma unroll
    for (int r = 0; r < 8; ++r) {
        size_t ro = (size_t)(rbase + r) * Nout;
        Cout[ro + col0] = c0[r] + bc0;
        Cout[ro + col1] = c1[r] + bc1;
    }
}

// -------------------- init per-(node,head) max / denom --------------------
__global__ void k_init_mden(unsigned* __restrict__ m_enc, float* __restrict__ den, int n) {
    int t = blockIdx.x * blockDim.x + threadIdx.x;
    if (t >= n) return;
    m_enc[t] = 0u;      // encodes -inf under the order-preserving map
    den[t]   = 0.f;
}

static __device__ __forceinline__ unsigned enc_f32(float f) {
    unsigned u = __float_as_uint(f);
    return (u & 0x80000000u) ? ~u : (u | 0x80000000u);
}
static __device__ __forceinline__ float dec_f32(unsigned u) {
    return __uint_as_float((u & 0x80000000u) ? (u & 0x7FFFFFFFu) : ~u);
}

// -------------------- attention logits + segment max --------------------
__global__ void k_alpha(const int* __restrict__ ei, const float* __restrict__ ea,
                        const float* __restrict__ q, const float* __restrict__ kk,
                        const float* __restrict__ We, float* __restrict__ alpha,
                        unsigned* __restrict__ m_enc, int H, int C, float scale) {
    __shared__ float sWe[3 * 128];
    int hc = H * C;
    for (int i = threadIdx.x; i < 3 * hc; i += blockDim.x) sWe[i] = We[i];
    __syncthreads();
    int t = blockIdx.x * blockDim.x + threadIdx.x;
    if (t >= EE * H) return;
    int e = t / H, hd = t % H;
    int src = ei[e], dst = ei[EE + e];
    float a0 = ea[e * 3 + 0], a1 = ea[e * 3 + 1], a2 = ea[e * 3 + 2];
    const float4* q4 = (const float4*)(q  + (size_t)dst * hc + hd * C);
    const float4* k4 = (const float4*)(kk + (size_t)src * hc + hd * C);
    float acc = 0.f;
    for (int cc = 0; cc < (C >> 2); ++cc) {
        float4 qv = q4[cc], kv = k4[cc];
        int j = hd * C + (cc << 2);
        float e0 = a0 * sWe[j + 0] + a1 * sWe[hc + j + 0] + a2 * sWe[2 * hc + j + 0];
        float e1 = a0 * sWe[j + 1] + a1 * sWe[hc + j + 1] + a2 * sWe[2 * hc + j + 1];
        float e2 = a0 * sWe[j + 2] + a1 * sWe[hc + j + 2] + a2 * sWe[2 * hc + j + 2];
        float e3 = a0 * sWe[j + 3] + a1 * sWe[hc + j + 3] + a2 * sWe[2 * hc + j + 3];
        acc += qv.x * (kv.x + e0) + qv.y * (kv.y + e1) + qv.z * (kv.z + e2) + qv.w * (kv.w + e3);
    }
    float al = acc * scale;
    alpha[t] = al;
    atomicMax(&m_enc[(size_t)dst * H + hd], enc_f32(al));
}

// -------------------- exp(alpha - max) + segment sum --------------------
__global__ void k_exp(const int* __restrict__ ei, float* __restrict__ alpha,
                      const unsigned* __restrict__ m_enc, float* __restrict__ den, int H) {
    int t = blockIdx.x * blockDim.x + threadIdx.x;
    if (t >= EE * H) return;
    int e = t / H, hd = t % H;
    int dst = ei[EE + e];
    float m = dec_f32(m_enc[(size_t)dst * H + hd]);
    float ex = __expf(alpha[t] - m);
    alpha[t] = ex;
    atomicAdd(&den[(size_t)dst * H + hd], ex);
}

// -------------------- weighted scatter of (v[src]+We*ea) onto skip output --------------------
__global__ void k_scatter(const int* __restrict__ ei, const float* __restrict__ ea,
                          const float* __restrict__ vv, const float* __restrict__ We,
                          const float* __restrict__ exv, const float* __restrict__ den,
                          float* __restrict__ hout, int H, int C, int dout, int concat) {
    __shared__ float sWe[3 * 128];
    int hc = H * C;
    for (int i = threadIdx.x; i < 3 * hc; i += blockDim.x) sWe[i] = We[i];
    __syncthreads();
    int t = blockIdx.x * blockDim.x + threadIdx.x;
    if (t >= EE * H) return;
    int e = t / H, hd = t % H;
    int src = ei[e], dst = ei[EE + e];
    float a = exv[t] / (den[(size_t)dst * H + hd] + 1e-16f);
    if (!concat) a *= (1.f / (float)H);   // mean over heads
    float a0 = ea[e * 3 + 0], a1 = ea[e * 3 + 1], a2 = ea[e * 3 + 2];
    const float4* v4 = (const float4*)(vv + (size_t)src * hc + hd * C);
    int colbase = concat ? hd * C : 0;
    float* orow = hout + (size_t)dst * dout + colbase;
    for (int cc = 0; cc < (C >> 2); ++cc) {
        float4 vvv = v4[cc];
        int j = hd * C + (cc << 2);
        float e0 = a0 * sWe[j + 0] + a1 * sWe[hc + j + 0] + a2 * sWe[2 * hc + j + 0];
        float e1 = a0 * sWe[j + 1] + a1 * sWe[hc + j + 1] + a2 * sWe[2 * hc + j + 1];
        float e2 = a0 * sWe[j + 2] + a1 * sWe[hc + j + 2] + a2 * sWe[2 * hc + j + 2];
        float e3 = a0 * sWe[j + 3] + a1 * sWe[hc + j + 3] + a2 * sWe[2 * hc + j + 3];
        int c0 = (cc << 2);
        atomicAdd(orow + c0 + 0, a * (vvv.x + e0));
        atomicAdd(orow + c0 + 1, a * (vvv.y + e1));
        atomicAdd(orow + c0 + 2, a * (vvv.z + e2));
        atomicAdd(orow + c0 + 3, a * (vvv.w + e3));
    }
}

// -------------------- layernorm + relu + optional residual; wave per node --------------------
__global__ void k_ln(const float* __restrict__ hacc, const float* __restrict__ g,
                     const float* __restrict__ b, float* __restrict__ h,
                     __bf16* __restrict__ hb, int dout, int residual) {
    int node = blockIdx.x * (blockDim.x >> 5) + (threadIdx.x >> 5);
    if (node >= NN) return;
    int lane = threadIdx.x & 31;
    int per = dout >> 5;                       // 4 (dout=128) or 1 (dout=32)
    const float* row = hacc + (size_t)node * dout;
    float vals[4];
    float s = 0.f;
    for (int i = 0; i < per; ++i) { vals[i] = row[lane + (i << 5)]; s += vals[i]; }
    for (int off = 16; off; off >>= 1) s += __shfl_xor(s, off, 32);
    float mu = s / (float)dout;
    float vsum = 0.f;
    for (int i = 0; i < per; ++i) { float d = vals[i] - mu; vsum += d * d; }
    for (int off = 16; off; off >>= 1) vsum += __shfl_xor(vsum, off, 32);
    float inv = rsqrtf(vsum / (float)dout + 1e-5f);
    float* hrow = h + (size_t)node * dout;
    __bf16* brow = hb + (size_t)node * dout;
    for (int i = 0; i < per; ++i) {
        int c = lane + (i << 5);
        float y = (vals[i] - mu) * inv * g[c] + b[c];
        y = fmaxf(y, 0.f);
        if (residual) y += hrow[c];            // read old h before overwrite (same thread)
        hrow[c] = y;
        brow[c] = (__bf16)y;
    }
}

// -------------------- fused edge MLP: relu([h[src],h[dst]]@W1+b1)@W2+b2 --------------------
// One wave = 16 edges.  Both 16-col accumulators live simultaneously; the 4 WMMAs
// alternate between the two independent chains to cover the hazard window.
__global__ void k_mlp(const int* __restrict__ ei, const __bf16* __restrict__ hb,
                      const __bf16* __restrict__ W1p, const float* __restrict__ b1,
                      const float* __restrict__ W2, const float* __restrict__ b2,
                      float* __restrict__ out) {
    __shared__ float sInter[4][16][32];
    __shared__ float sW2[32 * 3];
    __shared__ float sB2[3];
    int waveInBlk = threadIdx.x >> 5, lane = threadIdx.x & 31;
    int tile = blockIdx.x * 4 + waveInBlk;
    bool active = tile < (EE >> 4);
    if (active) {
        int erow = (tile << 4) + (lane & 15);
        int src = ei[erow], dst = ei[EE + erow];
        int kHalf = (lane < 16) ? 0 : 8;
        const __bf16* r0 = hb + (size_t)src * 32;   // K 0..31  = h[src]
        const __bf16* r1 = hb + (size_t)dst * 32;   // K 32..63 = h[dst]
        union { v16bf v; v8bf h[2]; } aS, aD;
        aS.h[0] = *(const v8bf*)(r0 + kHalf);  aS.h[1] = *(const v8bf*)(r0 + kHalf + 16);
        aD.h[0] = *(const v8bf*)(r1 + kHalf);  aD.h[1] = *(const v8bf*)(r1 + kHalf + 16);
        v16bf b00 = *(const v16bf*)(W1p + (((size_t)0 * 32) + lane) * 16);  // kt=0, nt=0
        v16bf b01 = *(const v16bf*)(W1p + (((size_t)1 * 32) + lane) * 16);  // kt=0, nt=1
        v16bf b10 = *(const v16bf*)(W1p + (((size_t)2 * 32) + lane) * 16);  // kt=1, nt=0
        v16bf b11 = *(const v16bf*)(W1p + (((size_t)3 * 32) + lane) * 16);  // kt=1, nt=1
        v8f c0 = {0.f, 0.f, 0.f, 0.f, 0.f, 0.f, 0.f, 0.f};
        v8f c1 = {0.f, 0.f, 0.f, 0.f, 0.f, 0.f, 0.f, 0.f};
        c0 = wmma_bf16(aS.v, b00, c0);
        c1 = wmma_bf16(aS.v, b01, c1);
        c0 = wmma_bf16(aD.v, b10, c0);
        c1 = wmma_bf16(aD.v, b11, c1);
        int nn = lane & 15;
        int rb = (lane >= 16) ? 8 : 0;
        float bc0 = b1[nn], bc1 = b1[16 + nn];
#pragma unroll
        for (int r = 0; r < 8; ++r) {
            sInter[waveInBlk][rb + r][nn]      = fmaxf(c0[r] + bc0, 0.f);
            sInter[waveInBlk][rb + r][16 + nn] = fmaxf(c1[r] + bc1, 0.f);
        }
    }
    for (int i = threadIdx.x; i < 96; i += blockDim.x) sW2[i] = W2[i];
    if (threadIdx.x < 3) sB2[threadIdx.x] = b2[threadIdx.x];
    __syncthreads();
    if (active && lane < 16) {
        int e = (tile << 4) + lane;
        float a0 = sB2[0], a1 = sB2[1], a2 = sB2[2];
        for (int c = 0; c < 32; ++c) {
            float hv = sInter[waveInBlk][lane][c];
            a0 += hv * sW2[c * 3 + 0];
            a1 += hv * sW2[c * 3 + 1];
            a2 += hv * sW2[c * 3 + 2];
        }
        out[e * 3 + 0] = a0;
        out[e * 3 + 1] = a1;
        out[e * 3 + 2] = a2;
    }
}

// ==================== host ====================
extern "C" void kernel_launch(void* const* d_in, const int* in_sizes, int n_in,
                              void* d_out, int out_size, void* d_ws, size_t ws_size,
                              hipStream_t stream) {
    (void)in_sizes; (void)n_in; (void)out_size; (void)ws_size;
    const int*   x   = (const int*)d_in[0];
    const int*   ei  = (const int*)d_in[1];
    const float* ea  = (const float*)d_in[2];
    const float* emb = (const float*)d_in[3];

    // workspace carve-out (~151 MB)
    char* ws = (char*)d_ws;
    size_t off = 0;
    auto carve = [&](size_t bytes) -> char* {
        char* p = ws + off;
        off += (bytes + 255) & ~(size_t)255;
        return p;
    };
    float*    hA    = (float*)carve((size_t)NN * EMBD * 4);   // current node features
    float*    hB    = (float*)carve((size_t)NN * EMBD * 4);   // skip-init + scatter accumulator
    __bf16*   hb    = (__bf16*)carve((size_t)NN * EMBD * 2);  // bf16 copy for WMMA A
    float*    qb    = (float*)carve((size_t)NN * EMBD * 4);
    float*    kb    = (float*)carve((size_t)NN * EMBD * 4);
    float*    vb    = (float*)carve((size_t)NN * EMBD * 4);
    float*    alpha = (float*)carve((size_t)EE * 4 * 4);
    unsigned* menc  = (unsigned*)carve((size_t)NN * 4 * 4);
    float*    den   = (float*)carve((size_t)NN * 4 * 4);
    __bf16*   pq    = (__bf16*)carve(128 * 128 * 2);
    __bf16*   pk    = (__bf16*)carve(128 * 128 * 2);
    __bf16*   pv    = (__bf16*)carve(128 * 128 * 2);
    __bf16*   psk   = (__bf16*)carve(128 * 128 * 2);
    __bf16*   w1p   = (__bf16*)carve(64 * 32 * 2);

    // h0 = emb[x]
    k_embed<<<(NN * EMBD + 255) / 256, 256, 0, stream>>>(x, emb, hA, hb);

    struct LC { int din, H, C, dout, concat, residual; };
    const LC cfg[3] = { {128, 4, 32, 128, 1, 1},
                       {128, 4, 32, 128, 1, 1},
                       {128, 1, 32,  32, 0, 0} };
    const float scale = 0.17677669529663687f;  // 1/sqrt(32)

    for (int l = 0; l < 3; ++l) {
        int base = 4 + 11 * l;
        const float* Wq  = (const float*)d_in[base + 0];
        const float* bq  = (const float*)d_in[base + 1];
        const float* Wk  = (const float*)d_in[base + 2];
        const float* bk  = (const float*)d_in[base + 3];
        const float* Wv  = (const float*)d_in[base + 4];
        const float* bv  = (const float*)d_in[base + 5];
        const float* We  = (const float*)d_in[base + 6];
        const float* Wsk = (const float*)d_in[base + 7];
        const float* bsk = (const float*)d_in[base + 8];
        const float* lng = (const float*)d_in[base + 9];
        const float* lnb = (const float*)d_in[base + 10];
        int din = cfg[l].din, H = cfg[l].H, C = cfg[l].C;
        int hc = H * C, dout = cfg[l].dout;

        // pack weights into WMMA B-fragment layout (bf16)
        int pe = din * hc, pbk = (pe + 255) / 256;
        k_pack_b<<<pbk, 256, 0, stream>>>(Wq,  pq,  din, hc);
        k_pack_b<<<pbk, 256, 0, stream>>>(Wk,  pk,  din, hc);
        k_pack_b<<<pbk, 256, 0, stream>>>(Wv,  pv,  din, hc);
        k_pack_b<<<pbk, 256, 0, stream>>>(Wsk, psk, din, dout);

        // q/k/v/skip projections via WMMA (16x32 tile per wave, dual accumulators)
        int tiles = (NN >> 4) * (hc >> 5);
        int gblk  = (tiles + 3) / 4;
        k_gemm_bf16<<<gblk, 128, 0, stream>>>(hb, pq,  bq,  qb, NN, din, hc);
        k_gemm_bf16<<<gblk, 128, 0, stream>>>(hb, pk,  bk,  kb, NN, din, hc);
        k_gemm_bf16<<<gblk, 128, 0, stream>>>(hb, pv,  bv,  vb, NN, din, hc);
        int tilesS = (NN >> 4) * (dout >> 5);
        k_gemm_bf16<<<(tilesS + 3) / 4, 128, 0, stream>>>(hb, psk, bsk, hB, NN, din, dout);

        // segment softmax + scatter
        int nmh = NN * H;
        k_init_mden<<<(nmh + 255) / 256, 256, 0, stream>>>(menc, den, nmh);
        int neh = EE * H, ablk = (neh + 255) / 256;
        k_alpha<<<ablk, 256, 0, stream>>>(ei, ea, qb, kb, We, alpha, menc, H, C, scale);
        k_exp<<<ablk, 256, 0, stream>>>(ei, alpha, menc, den, H);
        k_scatter<<<ablk, 256, 0, stream>>>(ei, ea, vb, We, alpha, den, hB, H, C, dout,
                                            cfg[l].concat);

        // layernorm + relu (+ residual), refresh f32 h and bf16 copy
        k_ln<<<(NN + 7) / 8, 256, 0, stream>>>(hB, lng, lnb, hA, hb, dout, cfg[l].residual);
    }

    // edge MLP head
    const float* W1 = (const float*)d_in[37];
    const float* b1 = (const float*)d_in[38];
    const float* W2 = (const float*)d_in[39];
    const float* b2 = (const float*)d_in[40];
    k_pack_b<<<(64 * 32 + 255) / 256, 256, 0, stream>>>(W1, w1p, 64, 32);
    int etiles = EE >> 4;
    k_mlp<<<(etiles + 3) / 4, 128, 0, stream>>>(ei, hb, w1p, b1, W2, b2, (float*)d_out);
}
